// LSTM_39041252721002
// MI455X (gfx1250) — matrix operand order repaired and checked by reading, not compiled, not run on hardware
//
#include <hip/hip_runtime.h>
#include <hip/hip_bf16.h>
#include <math.h>
#include <stdint.h>

typedef __attribute__((ext_vector_type(16))) __bf16 v16bf;
typedef __attribute__((ext_vector_type(8)))  __bf16 v8bf;
typedef __attribute__((ext_vector_type(8)))  float  v8f;
typedef __attribute__((ext_vector_type(4)))  int    v4i;

#define I_DIM 512
#define H_DIM 2048
#define O_DIM 512
#define T_DIM 4096
#define G4H   (4*H_DIM)   /* 8192 */

// gfx1250 async global->LDS copy path, guarded so the build never breaks.
#if defined(__HIP_DEVICE_COMPILE__) && \
    __has_builtin(__builtin_amdgcn_global_load_async_to_lds_b128) && \
    __has_builtin(__builtin_amdgcn_s_wait_asynccnt)
#define USE_ASYNC_LDS 1
typedef __attribute__((address_space(1))) v4i* gv4i_p;   // global (AS1) int4*
typedef __attribute__((address_space(3))) v4i* lv4i_p;   // LDS (AS3) int4*
#else
#define USE_ASYNC_LDS 0
#endif

// ---------------------------------------------------------------------------
// prep: fp32 -> bf16 cast (used for X and W_ih, row-major preserved)
// ---------------------------------------------------------------------------
__global__ void __launch_bounds__(256) cast_bf16(const float* __restrict__ src,
                                                 __bf16* __restrict__ dst, int n) {
    int i = blockIdx.x * 256 + threadIdx.x;
    if (i < n) dst[i] = (__bf16)src[i];
}

// ---------------------------------------------------------------------------
// prep: pack W_hh (8192x2048 fp32) into per-lane WMMA A-fragment order, bf16.
// Fragment element e (0..15) of lane l for tile (r,c):
//   M = 16r + (l&15),  K = 32c + 2*koff8 + e,  koff8 = (l<16?0:8)
// packed[((r*64+c)*32 + lane)*16 + e]  ->  64KB contiguous slab per row-tile r
// ---------------------------------------------------------------------------
__global__ void __launch_bounds__(256) pack_whh(const float* __restrict__ w,
                                                __bf16* __restrict__ p) {
    unsigned id   = blockIdx.x * 256u + threadIdx.x;   // 16,777,216 total
    unsigned e    = id & 15u;
    unsigned lane = (id >> 4) & 31u;
    unsigned c    = (id >> 9) & 63u;
    unsigned r    = id >> 15;
    unsigned koff8 = (lane < 16u) ? 0u : 8u;
    unsigned kl    = 2u * koff8 + e;
    unsigned row   = 16u * r + (lane & 15u);
    unsigned k     = 32u * c + kl;
    p[id] = (__bf16)w[(size_t)row * H_DIM + k];
}

// ---------------------------------------------------------------------------
// prep: zero h state (both parity buffers + fp32 copy) and barrier counter.
// Runs every launch -> deterministic under graph replay.
// ---------------------------------------------------------------------------
__global__ void __launch_bounds__(256) init_state(__bf16* __restrict__ hbf,
                                                  float* __restrict__ hf,
                                                  unsigned* __restrict__ ctr) {
    int t = threadIdx.x;
    for (int i = t; i < 2 * H_DIM; i += 256) hbf[i] = (__bf16)0.0f;
    for (int i = t; i < H_DIM; i += 256)     hf[i]  = 0.0f;
    if (t == 0) *ctr = 0u;
}

// ---------------------------------------------------------------------------
// Phase 1: XZ[t, n] = sum_k X[t,k] * W_ih[n,k] + (b_ih[n]+b_hh[n])
// One 16x16 output tile per wave; K=512 -> 16 v_wmma_f32_16x16x32_bf16,
// two independent accumulator chains.
// ---------------------------------------------------------------------------
__global__ void __launch_bounds__(256) gemm_xz(const __bf16* __restrict__ xb,
                                               const __bf16* __restrict__ wb,
                                               const float* __restrict__ bih,
                                               const float* __restrict__ bhh,
                                               float* __restrict__ xz) {
    const unsigned wave  = threadIdx.x >> 5;
    const unsigned lane  = threadIdx.x & 31u;
    const unsigned tile  = blockIdx.x * 8u + wave;   // 131072 tiles total
    const unsigned tt    = tile >> 9;                // 0..255 time-tiles
    const unsigned nt    = tile & 511u;              // 0..511 row-tiles
    const unsigned tbase = tt * 16u, nbase = nt * 16u;
    const unsigned l15   = lane & 15u;
    const unsigned hi    = lane >> 4;
    const unsigned koff8  = hi ? 8u : 0u;
    const unsigned koff16 = hi ? 16u : 0u;

    const __bf16* arow = xb + (size_t)(tbase + l15) * I_DIM;
    const __bf16* brow = wb + (size_t)(nbase + l15) * I_DIM;

    v8f acc0 = {}, acc1 = {};
#pragma unroll 4
    for (unsigned k = 0; k < I_DIM; k += 64) {
        {
            v8bf alo = *(const v8bf*)(arow + k + koff16);
            v8bf ahi = *(const v8bf*)(arow + k + koff16 + 8);
            v16bf a = __builtin_shufflevector(alo, ahi,
                        0,1,2,3,4,5,6,7,8,9,10,11,12,13,14,15);
            v16bf b = *(const v16bf*)(brow + k + koff16);
            acc0 = __builtin_amdgcn_wmma_f32_16x16x32_bf16(
                       false, a, false, b, (short)0, acc0, false, false);
        }
        {
            v8bf alo = *(const v8bf*)(arow + k + 32 + koff16);
            v8bf ahi = *(const v8bf*)(arow + k + 32 + koff16 + 8);
            v16bf a = __builtin_shufflevector(alo, ahi,
                        0,1,2,3,4,5,6,7,8,9,10,11,12,13,14,15);
            v16bf b = *(const v16bf*)(brow + k + 32 + koff16);
            acc1 = __builtin_amdgcn_wmma_f32_16x16x32_bf16(
                       false, a, false, b, (short)0, acc1, false, false);
        }
    }
    v8f acc = acc0 + acc1;
    const unsigned nrow = nbase + l15;
    const float bias = bih[nrow] + bhh[nrow];
#pragma unroll
    for (int d = 0; d < 8; ++d)
        xz[(size_t)(tbase + d + 8u * hi) * G4H + nrow] = acc[d] + bias;
}

// ---------------------------------------------------------------------------
// Phase 2: persistent recurrent kernel. 128 WGs x 128 threads (4 waves).
// WG g owns h-slice [16g,16g+16); wave w (= gate i/f/g/o) owns W_hh row-tile
// rtile = 128*w + g (rows 2048w+16g..+16).
// The WG's whole W_hh slab (4 x 64KB = 256KB) is staged into the 320KB WGP LDS
// ONCE; the 4096-step loop then runs entirely out of LDS (ds_load + v_wmma),
// zero global weight traffic. Mat-vec via WMMA: B = h chunk broadcast to all
// 16 columns -> every column of D is the mat-vec; lanes 0/16 carry all 16 row
// sums. 4 independent accumulators break the XDL RAW chain (1 wave/SIMD here).
// Double-buffered h (parity t&1), one grid barrier per step.
// ---------------------------------------------------------------------------
__global__ void __launch_bounds__(128) lstm_recur(const __bf16* __restrict__ whhp,
                                                  const float* __restrict__ xz,
                                                  __bf16* __restrict__ hbf,
                                                  float* __restrict__ hf32,
                                                  unsigned* __restrict__ ctr) {
    extern __shared__ __align__(32) __bf16 dyn_lds[];   // 256KB A + 4KB h
    __shared__ float z_lds[64];
    __shared__ float c_lds[16];
    __bf16* a_lds = dyn_lds;                    // [wave][c][lane][16] bf16
    __bf16* h_lds = dyn_lds + 4u * 32768u;      // 2048 bf16

    const unsigned tid    = threadIdx.x;
    const unsigned wave   = tid >> 5;      // gate index 0..3 (i,f,g,o)
    const unsigned lane   = tid & 31u;
    const unsigned g      = blockIdx.x;    // 0..127
    const unsigned hi     = lane >> 4;
    const unsigned koff16 = hi ? 16u : 0u;
    const unsigned rtile  = 128u * wave + g;                 // 0..511
    const unsigned nWG    = gridDim.x;

    // ---- one-time: stage this wave's 64KB W_hh slab into LDS ----
    {
        const v8bf* src = (const v8bf*)(whhp + (size_t)rtile * 32768u);
        v8bf*       dst = (v8bf*)(a_lds + (size_t)wave * 32768u);
        for (unsigned i = lane; i < 4096; i += 32) dst[i] = src[i];
    }
    if (tid < 16) c_lds[tid] = 0.0f;
    __syncthreads();

    const v16bf* aw = (const v16bf*)(a_lds + (size_t)wave * 32768u) + lane;

    for (unsigned t = 0; t < T_DIM; ++t) {
        // stage current h (bf16, 4KB) into LDS
        const __bf16* hcur = hbf + (size_t)(t & 1u) * H_DIM;
#if USE_ASYNC_LDS
        {
            const char* gsrc = (const char*)hcur;
            for (unsigned i = tid; i < 256; i += 128)
                __builtin_amdgcn_global_load_async_to_lds_b128(
                    (gv4i_p)(uintptr_t)(gsrc + 16u * i),
                    (lv4i_p)(uintptr_t)((char*)h_lds + 16u * i),
                    0, 0);
            __builtin_amdgcn_s_wait_asynccnt(0);
        }
#else
        for (unsigned i = tid; i < 256; i += 128)
            ((v8bf*)h_lds)[i] = ((const v8bf*)hcur)[i];
#endif
        __syncthreads();

        // mat-vec: 64 16x16x32 WMMAs, split over 4 independent accumulators
        v8f acc0 = {}, acc1 = {}, acc2 = {}, acc3 = {};
#pragma unroll 2
        for (unsigned c = 0; c < 64; c += 4) {
            v16bf a0 = aw[(size_t)(c + 0) * 32u];
            v16bf b0 = *(const v16bf*)(h_lds + 32u * (c + 0) + koff16);
            acc0 = __builtin_amdgcn_wmma_f32_16x16x32_bf16(
                       false, a0, false, b0, (short)0, acc0, false, false);
            v16bf a1 = aw[(size_t)(c + 1) * 32u];
            v16bf b1 = *(const v16bf*)(h_lds + 32u * (c + 1) + koff16);
            acc1 = __builtin_amdgcn_wmma_f32_16x16x32_bf16(
                       false, a1, false, b1, (short)0, acc1, false, false);
            v16bf a2 = aw[(size_t)(c + 2) * 32u];
            v16bf b2 = *(const v16bf*)(h_lds + 32u * (c + 2) + koff16);
            acc2 = __builtin_amdgcn_wmma_f32_16x16x32_bf16(
                       false, a2, false, b2, (short)0, acc2, false, false);
            v16bf a3 = aw[(size_t)(c + 3) * 32u];
            v16bf b3 = *(const v16bf*)(h_lds + 32u * (c + 3) + koff16);
            acc3 = __builtin_amdgcn_wmma_f32_16x16x32_bf16(
                       false, a3, false, b3, (short)0, acc3, false, false);
        }
        v8f acc = (acc0 + acc1) + (acc2 + acc3);

        // lanes 0 / 16 hold rows d / d+8 of the tile in acc[d]
        if ((lane & 15u) == 0u) {
            const size_t zb = (size_t)t * G4H + (size_t)rtile * 16u + 8u * hi;
#pragma unroll
            for (int d = 0; d < 8; ++d)
                z_lds[wave * 16u + 8u * hi + d] = acc[d] + xz[zb + d];
        }
        __syncthreads();

        if (tid < 16) {
            float zi = z_lds[tid];
            float zf = z_lds[16 + tid];
            float zg = z_lds[32 + tid];
            float zo = z_lds[48 + tid];
            float ig = 1.0f / (1.0f + __expf(-zi));
            float fg = 1.0f / (1.0f + __expf(-zf));
            float gg = tanhf(zg);
            float og = 1.0f / (1.0f + __expf(-zo));
            float cc = fg * c_lds[tid] + ig * gg;
            c_lds[tid] = cc;
            float hh = og * tanhf(cc);
            unsigned idx = g * 16u + tid;
            hbf[(size_t)((t + 1u) & 1u) * H_DIM + idx] = (__bf16)hh;
            hf32[idx] = hh;
        }
        __syncthreads();

        // grid-wide barrier (release stores, acquire spin)
        if (tid == 0) {
            __threadfence();
            __hip_atomic_fetch_add(ctr, 1u, __ATOMIC_RELEASE,
                                   __HIP_MEMORY_SCOPE_AGENT);
            const unsigned target = (t + 1u) * nWG;
            while (__hip_atomic_load(ctr, __ATOMIC_ACQUIRE,
                                     __HIP_MEMORY_SCOPE_AGENT) < target) {
                __builtin_amdgcn_s_sleep(2);
            }
        }
        __syncthreads();
    }
}

// ---------------------------------------------------------------------------
// Phase 3: out = W_lin @ h_T + b_lin   (512x2048, trivial, fp32)
// ---------------------------------------------------------------------------
__global__ void __launch_bounds__(256) final_linear(const float* __restrict__ wlin,
                                                    const float* __restrict__ blin,
                                                    const float* __restrict__ hf,
                                                    float* __restrict__ out) {
    int o = blockIdx.x * 256 + threadIdx.x;
    if (o >= O_DIM) return;
    const float* wr = wlin + (size_t)o * H_DIM;
    float s = 0.0f;
    for (int k = 0; k < H_DIM; k += 4)
        s += wr[k] * hf[k] + wr[k+1] * hf[k+1]
           + wr[k+2] * hf[k+2] + wr[k+3] * hf[k+3];
    out[o] = s + blin[o];
}

// ---------------------------------------------------------------------------
extern "C" void kernel_launch(void* const* d_in, const int* in_sizes, int n_in,
                              void* d_out, int out_size, void* d_ws, size_t ws_size,
                              hipStream_t stream) {
    (void)in_sizes; (void)n_in; (void)out_size; (void)ws_size;
    const float* x    = (const float*)d_in[0];   // (T, I)
    const float* wih  = (const float*)d_in[1];   // (4H, I)
    const float* whh  = (const float*)d_in[2];   // (4H, H)
    const float* bih  = (const float*)d_in[3];   // (4H,)
    const float* bhh  = (const float*)d_in[4];   // (4H,)
    const float* wlin = (const float*)d_in[5];   // (O, H)
    const float* blin = (const float*)d_in[6];   // (O,)
    float* out = (float*)d_out;

    char* ws = (char*)d_ws;
    size_t off = 0;
    auto alloc = [&](size_t bytes) -> void* {
        void* p = ws + off;
        off = (off + bytes + 255) & ~(size_t)255;
        return p;
    };
    __bf16*   xb   = (__bf16*)  alloc((size_t)T_DIM * I_DIM * 2);   //   4 MB
    __bf16*   wihb = (__bf16*)  alloc((size_t)G4H * I_DIM * 2);     //   8 MB
    __bf16*   whhp = (__bf16*)  alloc((size_t)G4H * H_DIM * 2);     //  32 MB
    float*    xz   = (float*)   alloc((size_t)T_DIM * G4H * 4);     // 128 MB
    __bf16*   hbf  = (__bf16*)  alloc((size_t)2 * H_DIM * 2);
    float*    hf   = (float*)   alloc((size_t)H_DIM * 4);
    unsigned* ctr  = (unsigned*)alloc(256);

    cast_bf16<<<(T_DIM * I_DIM) / 256, 256, 0, stream>>>(x, xb, T_DIM * I_DIM);
    cast_bf16<<<(G4H * I_DIM) / 256, 256, 0, stream>>>(wih, wihb, G4H * I_DIM);
    pack_whh<<<(G4H * H_DIM) / 256, 256, 0, stream>>>(whh, whhp);
    init_state<<<1, 256, 0, stream>>>(hbf, hf, ctr);

    gemm_xz<<<16384, 256, 0, stream>>>(xb, wihb, bih, bhh, xz);

    // 256KB W_hh slab + 4KB h buffer staged in dynamic LDS (320KB/WGP)
    const size_t recur_lds = (size_t)4 * 32768 * sizeof(__bf16) + H_DIM * sizeof(__bf16);
    lstm_recur<<<128, 128, recur_lds, stream>>>(whhp, xz, hbf, hf, ctr);

    final_linear<<<2, 256, 0, stream>>>(wlin, blin, hf, out);
}
